// GPT2_11768210391393
// MI455X (gfx1250) — matrix operand order repaired and checked by reading, not compiled, not run on hardware
//
#include <hip/hip_runtime.h>
#include <hip/hip_bf16.h>
#include <math.h>

// Problem constants (from reference setup_inputs)
#define BB 2
#define SS 512
#define DD 2560
#define HH 32
#define DHD 80
#define FF 10240
#define NL 2
#define MROWS (BB * SS) // 1024

typedef __attribute__((ext_vector_type(16))) __bf16 v16bf;
typedef __attribute__((ext_vector_type(8)))  __bf16 v8bf;
typedef __attribute__((ext_vector_type(8)))  float  v8f;

__device__ __forceinline__ __bf16 to_bf16(float f) {
    // round-to-nearest-even fp32 -> bf16
    unsigned u = __builtin_bit_cast(unsigned, f);
    u += 0x7FFFu + ((u >> 16) & 1u);
    unsigned short h = (unsigned short)(u >> 16);
    return __builtin_bit_cast(__bf16, h);
}

// ---------------------------------------------------------------------------
// Embedding gather: X[row,:] = wte[ids[row],:] + wpe[row % S,:]
// ---------------------------------------------------------------------------
__global__ __launch_bounds__(256)
void embed_kernel(const int* __restrict__ ids, const float* __restrict__ wte,
                  const float* __restrict__ wpe, float* __restrict__ X) {
    const int row = blockIdx.x;
    const int tid = threadIdx.x;
    const int s = row % SS;
    const int id = ids[row];
    const float* wt = wte + (size_t)id * DD;
    const float* wp = wpe + (size_t)s * DD;
    float* x = X + (size_t)row * DD;
    for (int d = tid; d < DD; d += 256) x[d] = wt[d] + wp[d];
}

// ---------------------------------------------------------------------------
// LayerNorm: one block (256 thr) per row of D=2560
// ---------------------------------------------------------------------------
__global__ __launch_bounds__(256)
void layernorm_kernel(const float* __restrict__ X, const float* __restrict__ g,
                      const float* __restrict__ b, float* __restrict__ Y) {
    __shared__ float s1[256];
    __shared__ float s2[256];
    const int row = blockIdx.x;
    const int tid = threadIdx.x;
    const float* x = X + (size_t)row * DD;
    float sum = 0.f, sq = 0.f;
    for (int d = tid; d < DD; d += 256) { float v = x[d]; sum += v; sq += v * v; }
    s1[tid] = sum; s2[tid] = sq;
    __syncthreads();
    for (int off = 128; off; off >>= 1) {
        if (tid < off) { s1[tid] += s1[tid + off]; s2[tid] += s2[tid + off]; }
        __syncthreads();
    }
    const float mean = s1[0] * (1.0f / DD);
    float var = s2[0] * (1.0f / DD) - mean * mean;
    var = var < 0.f ? 0.f : var;
    const float inv = rsqrtf(var + 1e-12f);
    float* y = Y + (size_t)row * DD;
    for (int d = tid; d < DD; d += 256)
        y[d] = (x[d] - mean) * inv * g[d] + b[d];
}

// ---------------------------------------------------------------------------
// bf16 WMMA GEMM: C = act(A[MxK] @ B[KxN] + bias) (+ residual)
// 128x128 block tile, 256 thr = 8 waves (2 x 4). Each wave: 64x32 = 4x2 wmma
// fragments -> 8 v_wmma per K-step per wave. BK=32 (one wmma consumes it).
// ---------------------------------------------------------------------------
__global__ __launch_bounds__(256)
void gemm_bf16_wmma(const float* __restrict__ A, const float* __restrict__ Bw,
                    const float* __restrict__ bias, const float* __restrict__ Res,
                    float* __restrict__ C, int M, int N, int Kd, int act) {
    __shared__ __align__(64) __bf16 ldsA[128 * 32];  // [m][k], row stride 32
    __shared__ __align__(64) __bf16 ldsB[128 * 32];  // transposed: [n][k], row stride 32

    const int tid  = threadIdx.x;
    const int lane = tid & 31;
    const int w    = tid >> 5;           // 0..7
    const int wm   = (w & 1) * 64;       // wave m offset inside 128x128 tile
    const int wn   = (w >> 1) * 32;      // wave n offset inside 128x128 tile
    const int half = lane >> 4;          // 0 / 1 (lane group)
    const int l16  = lane & 15;
    const int m0   = blockIdx.y * 128;
    const int n0   = blockIdx.x * 128;

    v8f acc[4][2];
#pragma unroll
    for (int i = 0; i < 4; ++i)
#pragma unroll
        for (int j = 0; j < 2; ++j)
#pragma unroll
            for (int e = 0; e < 8; ++e) acc[i][j][e] = 0.f;

    for (int k0 = 0; k0 < Kd; k0 += 32) {
        // --- stage A tile (128 rows x 32 k), fp32 -> bf16 ---
#pragma unroll
        for (int it = 0; it < 4; ++it) {
            const int idx = it * 256 + tid;       // 0..1023 float4 slots
            const int row = idx >> 3;             // 0..127
            const int c4  = (idx & 7) * 4;        // 0,4,..,28
            const float4 f = *(const float4*)&A[(size_t)(m0 + row) * Kd + k0 + c4];
            __bf16* dst = &ldsA[row * 32 + c4];
            dst[0] = to_bf16(f.x); dst[1] = to_bf16(f.y);
            dst[2] = to_bf16(f.z); dst[3] = to_bf16(f.w);
        }
        // --- stage B tile (32 k x 128 n), transposed into [n][k] ---
#pragma unroll
        for (int it = 0; it < 4; ++it) {
            const int idx = it * 256 + tid;       // 0..1023 float4 slots
            const int r   = idx >> 5;             // k row 0..31
            const int c4  = (idx & 31) * 4;       // n offset 0..124
            const float4 f = *(const float4*)&Bw[(size_t)(k0 + r) * N + n0 + c4];
            ldsB[(c4 + 0) * 32 + r] = to_bf16(f.x);
            ldsB[(c4 + 1) * 32 + r] = to_bf16(f.y);
            ldsB[(c4 + 2) * 32 + r] = to_bf16(f.z);
            ldsB[(c4 + 3) * 32 + r] = to_bf16(f.w);
        }
        // prefetch next K-tile into L2/L0 while WMMA crunches this one
        if (k0 + 32 < Kd) {
            const int idxp = tid;
            const int rowp = idxp >> 3;
            __builtin_prefetch(&A[(size_t)(m0 + rowp * 4) * Kd + k0 + 32], 0, 3);
            __builtin_prefetch(&Bw[(size_t)(k0 + 32 + (idxp >> 5)) * N + n0 + (idxp & 31) * 4], 0, 3);
        }
        __syncthreads();

        // --- build fragments (contiguous k order, same for A and B) ---
        v16bf afrag[4], bfrag[2];
#pragma unroll
        for (int i = 0; i < 4; ++i) {
            const int row = wm + i * 16 + l16;    // M = lane%16
            // lanes 0-15 hold K 0-15; lanes 16-31 hold K 16-31 (contiguous)
            afrag[i] = *(const v16bf*)&ldsA[row * 32 + half * 16];
        }
#pragma unroll
        for (int j = 0; j < 2; ++j) {
            const int col = wn + j * 16 + l16;    // N = lane%16
            // lanes 0-15 hold K 0-15; lanes 16-31 hold K 16-31 (contiguous)
            bfrag[j] = *(const v16bf*)&ldsB[col * 32 + half * 16];
        }
#pragma unroll
        for (int i = 0; i < 4; ++i)
#pragma unroll
            for (int j = 0; j < 2; ++j)
                acc[i][j] = __builtin_amdgcn_wmma_f32_16x16x32_bf16(
                    false, afrag[i], false, bfrag[j], (short)0, acc[i][j], false, false);
        __syncthreads();
    }

    // --- epilogue: bias (+gelu) (+residual). C/D layout: n=lane%16, m=v+8*(lane/16)
#pragma unroll
    for (int i = 0; i < 4; ++i) {
#pragma unroll
        for (int j = 0; j < 2; ++j) {
            const int mrow = m0 + wm + i * 16 + half * 8;
            const int ncol = n0 + wn + j * 16 + l16;
            const float bv = bias[ncol];
#pragma unroll
            for (int v = 0; v < 8; ++v) {
                float r = acc[i][j][v] + bv;
                if (act) // exact GELU
                    r = 0.5f * r * (1.0f + erff(r * 0.70710678118654752f));
                const size_t off = (size_t)(mrow + v) * N + ncol;
                if (Res) r += Res[off];
                C[off] = r;
            }
        }
    }
}

// ---------------------------------------------------------------------------
// Fused causal attention: one block per (b, h, q-row). scores -> softmax -> ctx
// ---------------------------------------------------------------------------
__global__ __launch_bounds__(256)
void attn_kernel(const float* __restrict__ Q, const float* __restrict__ Kt,
                 const float* __restrict__ V, float* __restrict__ CTX) {
    __shared__ float qs[DHD];
    __shared__ float sc[SS];
    __shared__ float red[256];
    const int qi = blockIdx.x, h = blockIdx.y, b = blockIdx.z;
    const int tid = threadIdx.x;
    const size_t qoff = (size_t)(b * SS + qi) * DD + h * DHD;
    if (tid < DHD) qs[tid] = Q[qoff + tid];
    __syncthreads();
    const float scale = 0.11180339887498949f; // 1/sqrt(80)
    for (int kk = tid; kk < SS; kk += 256) {
        float s;
        if (kk <= qi) {
            const float* kv = Kt + (size_t)(b * SS + kk) * DD + h * DHD;
            float a = 0.f;
#pragma unroll 8
            for (int d = 0; d < DHD; ++d) a += qs[d] * kv[d];
            s = a * scale;
        } else {
            s = -1.0e12f;
        }
        sc[kk] = s;
    }
    __syncthreads();
    // row max
    float m = fmaxf(sc[tid], sc[tid + 256]);
    red[tid] = m; __syncthreads();
    for (int off = 128; off; off >>= 1) {
        if (tid < off) red[tid] = fmaxf(red[tid], red[tid + off]);
        __syncthreads();
    }
    const float rowmax = red[0];
    __syncthreads();
    // exp + sum
    const float e1 = __expf(sc[tid] - rowmax);
    const float e2 = __expf(sc[tid + 256] - rowmax);
    sc[tid] = e1; sc[tid + 256] = e2;
    red[tid] = e1 + e2; __syncthreads();
    for (int off = 128; off; off >>= 1) {
        if (tid < off) red[tid] += red[tid + off];
        __syncthreads();
    }
    const float rsum = 1.0f / red[0];
    __syncthreads();
    // ctx = probs @ V (80 outputs)
    for (int d = tid; d < DHD; d += 256) {
        const float* vb = V + (size_t)(b * SS) * DD + h * DHD + d;
        float a = 0.f;
        for (int kk = 0; kk < SS; ++kk) a += sc[kk] * vb[(size_t)kk * DD];
        CTX[qoff + d] = a * rsum;
    }
}

// ---------------------------------------------------------------------------
// Host launcher
// ---------------------------------------------------------------------------
extern "C" void kernel_launch(void* const* d_in, const int* in_sizes, int n_in,
                              void* d_out, int out_size, void* d_ws, size_t ws_size,
                              hipStream_t stream) {
    const int*   ids  = (const int*)d_in[0];
    const float* wte  = (const float*)d_in[1];
    const float* wpe  = (const float*)d_in[2];
    const float* ln0g = (const float*)d_in[3];
    const float* ln0b = (const float*)d_in[4];
    const float* wq   = (const float*)d_in[5];
    const float* bq   = (const float*)d_in[6];
    const float* wk   = (const float*)d_in[7];
    const float* bk   = (const float*)d_in[8];
    const float* wv   = (const float*)d_in[9];
    const float* bv   = (const float*)d_in[10];
    const float* wo   = (const float*)d_in[11];
    const float* bo   = (const float*)d_in[12];
    const float* ln1g = (const float*)d_in[13];
    const float* ln1b = (const float*)d_in[14];
    const float* w1   = (const float*)d_in[15];
    const float* b1   = (const float*)d_in[16];
    const float* w2   = (const float*)d_in[17];
    const float* b2   = (const float*)d_in[18];
    (void)in_sizes; (void)n_in; (void)out_size; (void)ws_size;

    const size_t SZ = (size_t)MROWS * DD;   // 2.62M floats
    float* X   = (float*)d_ws;              // residual stream
    float* HN  = X + SZ;                    // layernorm output
    float* Qb  = HN + SZ;
    float* Kb  = Qb + SZ;
    float* Vb  = Kb + SZ;
    float* CT  = Vb + SZ;
    float* MID = CT + SZ;                   // MROWS * FF (4*SZ)

    embed_kernel<<<MROWS, 256, 0, stream>>>(ids, wte, wpe, X);

    const dim3 gD(DD / 128, MROWS / 128);   // (20, 8)
    const dim3 gF(FF / 128, MROWS / 128);   // (80, 8)
    for (int l = 0; l < NL; ++l) {
        const float* wq_l = wq + (size_t)l * DD * DD;
        const float* wk_l = wk + (size_t)l * DD * DD;
        const float* wv_l = wv + (size_t)l * DD * DD;
        const float* wo_l = wo + (size_t)l * DD * DD;
        const float* w1_l = w1 + (size_t)l * DD * FF;
        const float* w2_l = w2 + (size_t)l * FF * DD;
        const float* bq_l = bq + (size_t)l * DD;
        const float* bk_l = bk + (size_t)l * DD;
        const float* bv_l = bv + (size_t)l * DD;
        const float* bo_l = bo + (size_t)l * DD;
        const float* b1_l = b1 + (size_t)l * FF;
        const float* b2_l = b2 + (size_t)l * DD;

        layernorm_kernel<<<MROWS, 256, 0, stream>>>(X, ln0g + (size_t)l * DD, ln0b + (size_t)l * DD, HN);
        gemm_bf16_wmma<<<gD, 256, 0, stream>>>(HN, wq_l, bq_l, nullptr, Qb, MROWS, DD, DD, 0);
        gemm_bf16_wmma<<<gD, 256, 0, stream>>>(HN, wk_l, bk_l, nullptr, Kb, MROWS, DD, DD, 0);
        gemm_bf16_wmma<<<gD, 256, 0, stream>>>(HN, wv_l, bv_l, nullptr, Vb, MROWS, DD, DD, 0);
        attn_kernel<<<dim3(SS, HH, BB), 256, 0, stream>>>(Qb, Kb, Vb, CT);
        // x = x + ctx @ wo + bo  (in-place residual: each element read once then written)
        gemm_bf16_wmma<<<gD, 256, 0, stream>>>(CT, wo_l, bo_l, X, X, MROWS, DD, DD, 0);
        layernorm_kernel<<<MROWS, 256, 0, stream>>>(X, ln1g + (size_t)l * DD, ln1b + (size_t)l * DD, HN);
        gemm_bf16_wmma<<<gF, 256, 0, stream>>>(HN, w1_l, b1_l, nullptr, MID, MROWS, FF, DD, 1);
        float* xo = (l == NL - 1) ? (float*)d_out : X;
        gemm_bf16_wmma<<<gD, 256, 0, stream>>>(MID, w2_l, b2_l, X, xo, MROWS, DD, FF, 0);
    }
}
